// LightweightIndustrialDiffusion_58866821759338
// MI455X (gfx1250) — matrix operand order repaired and checked by reading, not compiled, not run on hardware
//
#include <hip/hip_runtime.h>

// MI455X / gfx1250, wave32. WMMA f32<=f16 16x16x32 for all dense GEMMs.
// All WMMA operands pre-converted/transposed to f16 so fragments load as
// contiguous v8h (b128) vectors. Fused bilinear+MLP head keeps the 134MB
// "pair" tensor in LDS.

typedef __attribute__((ext_vector_type(16))) _Float16 v16h;
typedef __attribute__((ext_vector_type(8)))  _Float16 v8h;
typedef __attribute__((ext_vector_type(8)))  float    v8f;
typedef _Float16 h16;

#define NN   1024      // N nodes
#define HH   128       // hidden
#define EE   65536     // edges
#define NPG  256
#define NL   6

__device__ __forceinline__ float siluf(float x){ return x / (1.0f + __expf(-x)); }
__device__ __forceinline__ float geluf(float x){ return 0.5f * x * (1.0f + erff(x * 0.70710678118654752f)); }
__device__ __forceinline__ unsigned f2mono(float f){ unsigned u = __float_as_uint(f); return (u >> 31) ? ~u : (u | 0x80000000u); }
__device__ __forceinline__ float mono2f(unsigned e){ unsigned u = (e >> 31) ? (e & 0x7fffffffu) : ~e; return __uint_as_float(u); }

// A fragment: rows = M (lane&15), two contiguous 8-half runs K{0..7},{16..23}
// (+8 for upper half-wave), per ISA 7.12.2 16-bit A layout.
__device__ __forceinline__ v16h load_afrag(const h16* rowbase /* already + hf*8 + kb */) {
    v8h c0 = *(const v8h*)(rowbase);
    v8h c1 = *(const v8h*)(rowbase + 16);
    return __builtin_shufflevector(c0, c1, 0,1,2,3,4,5,6,7,8,9,10,11,12,13,14,15);
}
// B fragment from [N,K]-transposed f16: per lane two 8-half runs K{0..7},{16..23} (+8 for upper half-wave).
__device__ __forceinline__ v16h load_bfrag(const h16* p /* Bt + n*K + kb + hf*8 */) {
    v8h c0 = *(const v8h*)(p);
    v8h c1 = *(const v8h*)(p + 16);
    return __builtin_shufflevector(c0, c1, 0,1,2,3,4,5,6,7,8,9,10,11,12,13,14,15);
}

// ---------------- setup: timestep embedding + per-layer modulation ----------
__global__ void setup_kernel(const int* tptr,
                             const float* __restrict__ Wt1, const float* __restrict__ bt1,
                             const float* __restrict__ Wt2, const float* __restrict__ bt2,
                             const float* __restrict__ Wa,  const float* __restrict__ ba,
                             float* __restrict__ ss)
{
    __shared__ float semb[256];
    __shared__ float sa1[128];
    __shared__ float stmod[128];
    int t = threadIdx.x; // 512 threads
    float tv = (float)tptr[0];
    if (t < 128) {
        float fr = __expf(-logf(10000.0f) * (float)t / 128.0f);
        float ta = tv * fr;
        semb[t] = cosf(ta);
        semb[128 + t] = sinf(ta);
    }
    __syncthreads();
    if (t < 128) {
        float acc = bt1[t];
        for (int f = 0; f < 256; ++f) acc += semb[f] * Wt1[f * HH + t];
        sa1[t] = siluf(acc);
    }
    __syncthreads();
    if (t < 128) {
        float acc = bt2[t];
        for (int k = 0; k < 128; ++k) acc += sa1[k] * Wt2[k * HH + t];
        stmod[t] = siluf(acc);
    }
    __syncthreads();
    for (int l = 0; l < NL; ++l) {
        float acc = ba[l * 512 + t];
        for (int k = 0; k < 128; ++k) acc += stmod[k] * Wa[(l * 128 + k) * 512 + t];
        ss[l * 512 + t] = acc;
    }
}

// ---------------- h0 = x @ Wn + bn ----------------
__global__ void input_proj_kernel(const float* __restrict__ x, const float* __restrict__ Wn,
                                  const float* __restrict__ bn, float* __restrict__ h)
{
    int n = blockIdx.x, j = threadIdx.x;
    float acc = bn[j];
    for (int k = 0; k < 5; ++k) acc += x[n * 5 + k] * Wn[k * HH + j];
    h[n * HH + j] = acc;
}

// ---------------- edge_attr ----------------
__global__ void edge_kernel(const int* __restrict__ ei, const float* __restrict__ tmat,
                            const float* __restrict__ Wee, const float* __restrict__ bee,
                            float* __restrict__ ea)
{
    int e = blockIdx.x * blockDim.x + threadIdx.x;
    if (e >= EE) return;
    int s = ei[e], d = ei[EE + e];
    ea[e] = tmat[s * NN + d] * Wee[0] + bee[0];
}

// ---------------- transpose+convert weights: Bt[n*K+k] = (h16)B[k*N+n] ------
__global__ void tconv_kernel(const float* __restrict__ B, h16* __restrict__ Bt,
                             int K, int N)
{
    int idx = blockIdx.x * blockDim.x + threadIdx.x;
    if (idx >= K * N) return;
    int k = idx / N, n = idx - k * N;
    Bt[n * K + k] = (h16)B[idx];
}

// WbT[o][l][k] = (h16)Wb[o][k][l]
__global__ void tconv_wb_kernel(const float* __restrict__ Wb, h16* __restrict__ WbT)
{
    int idx = blockIdx.x * blockDim.x + threadIdx.x; // 128*128*128
    int o = idx >> 14;
    int r = idx & 16383;
    int k = r >> 7, l = r & 127;
    WbT[(o << 14) + (l << 7) + k] = (h16)Wb[idx];
}

// ---------------- LayerNorm + adaLN modulation -> f16 ----------------
__global__ void ln_mod_kernel(const float* __restrict__ h, const float* __restrict__ sh,
                              const float* __restrict__ sc, h16* __restrict__ hn16)
{
    __shared__ float red[128];
    int n = blockIdx.x, t = threadIdx.x;
    float v = h[n * HH + t];
    red[t] = v; __syncthreads();
    for (int s = 64; s > 0; s >>= 1) { if (t < s) red[t] += red[t + s]; __syncthreads(); }
    float mu = red[0] * (1.0f / 128.0f);
    __syncthreads();
    float d = v - mu;
    red[t] = d * d; __syncthreads();
    for (int s = 64; s > 0; s >>= 1) { if (t < s) red[t] += red[t + s]; __syncthreads(); }
    float var = red[0] * (1.0f / 128.0f);
    hn16[n * HH + t] = (h16)(d * rsqrtf(var + 1e-5f) * (1.0f + sc[t]) + sh[t]);
}

// ---------------- generic WMMA GEMM: C = act(A@Bt^T + bias) ----------------
// A f16 [M,K] row-major, Bt f16 [N,K] (pre-transposed), C f32 [M,ldc].
__global__ void gemm_wmma_h16(const h16* __restrict__ A, int lda,
                              const h16* __restrict__ Bt,
                              const float* __restrict__ bias,
                              float* __restrict__ C, int ldc,
                              int K, int act)
{
    int lane = threadIdx.x & 31;
    int ml = lane & 15;
    int hf = (lane >= 16);
    int tm = blockIdx.x * 16;
    int tn = blockIdx.y * 16;
    v8f acc;
    float bv = bias[tn + ml];
    #pragma unroll
    for (int r = 0; r < 8; ++r) acc[r] = bv;
    const h16* Arow = A + (size_t)(tm + ml) * lda + (hf ? 8 : 0);
    const h16* Brow = Bt + (size_t)(tn + ml) * K + (hf ? 8 : 0);
    for (int kb = 0; kb < K; kb += 32) {
        v16h af = load_afrag(Arow + kb);
        v16h bf = load_bfrag(Brow + kb);
        acc = __builtin_amdgcn_wmma_f32_16x16x32_f16(false, af, false, bf, (short)0, acc, false, false);
    }
    #pragma unroll
    for (int r = 0; r < 8; ++r) {
        int m = tm + r + (hf ? 8 : 0);
        float v = acc[r];
        if (act == 1) v = geluf(v);
        C[(size_t)m * ldc + tn + ml] = v;
    }
}

// ---------------- attention: per-layer segment softmax over edges ----------
__global__ void layer_init_kernel(float* __restrict__ attn, unsigned* __restrict__ mseg,
                                  float* __restrict__ ssum)
{
    int i = blockIdx.x * blockDim.x + threadIdx.x;
    if (i < NN * HH) attn[i] = 0.0f;
    if (i < NN * 4) { mseg[i] = 0u; ssum[i] = 0.0f; }
}

__global__ void attn_alpha_kernel(const int* __restrict__ ei, const float* __restrict__ ea,
                                  const float* __restrict__ q, const float* __restrict__ k,
                                  const float* __restrict__ WeL,
                                  float* __restrict__ alpha, unsigned* __restrict__ mseg)
{
    int gid = blockIdx.x * blockDim.x + threadIdx.x;
    if (gid >= EE * 4) return;
    int e = gid >> 2, hh = gid & 3;
    int s = ei[e], d = ei[EE + e];
    float eav = ea[e];
    int co = hh * 32;
    float acc = 0.0f;
    #pragma unroll 8
    for (int c = 0; c < 32; ++c) {
        float kj = k[s * HH + co + c] + eav * WeL[co + c];
        acc += q[d * HH + co + c] * kj;
    }
    acc *= 0.17677669529663687f; // 1/sqrt(32)
    alpha[e * 4 + hh] = acc;
    atomicMax(&mseg[d * 4 + hh], f2mono(acc));
}

__global__ void attn_exp_kernel(const int* __restrict__ ei, float* __restrict__ alpha,
                                const unsigned* __restrict__ mseg, float* __restrict__ ssum)
{
    int gid = blockIdx.x * blockDim.x + threadIdx.x;
    if (gid >= EE * 4) return;
    int e = gid >> 2, hh = gid & 3;
    int d = ei[EE + e];
    float m = mono2f(mseg[d * 4 + hh]);
    float a = __expf(alpha[e * 4 + hh] - m);
    alpha[e * 4 + hh] = a;
    atomicAdd(&ssum[d * 4 + hh], a);
}

__global__ void attn_scatter_kernel(const int* __restrict__ ei, const float* __restrict__ ea,
                                    const float* __restrict__ v, const float* __restrict__ WeL,
                                    const float* __restrict__ alpha, const float* __restrict__ ssum,
                                    float* __restrict__ attn)
{
    int gid = blockIdx.x * blockDim.x + threadIdx.x;
    if (gid >= EE * 4) return;
    int e = gid >> 2, hh = gid & 3;
    int s = ei[e], d = ei[EE + e];
    float eav = ea[e];
    float w = alpha[e * 4 + hh] / (ssum[d * 4 + hh] + 1e-16f);
    int co = hh * 32;
    #pragma unroll 8
    for (int c = 0; c < 32; ++c)
        atomicAdd(&attn[d * HH + co + c], (v[s * HH + co + c] + eav * WeL[co + c]) * w);
}

// ---------------- elementwise helpers ----------------
__global__ void add2_kernel(float* __restrict__ h, const float* __restrict__ a,
                            const float* __restrict__ b)
{ int i = blockIdx.x * blockDim.x + threadIdx.x; if (i < NN * HH) h[i] += a[i] + b[i]; }

__global__ void add1_kernel(float* __restrict__ h, const float* __restrict__ a)
{ int i = blockIdx.x * blockDim.x + threadIdx.x; if (i < NN * HH) h[i] += a[i]; }

__global__ void cvt_kernel(const float* __restrict__ src, h16* __restrict__ dst, int n)
{ int i = blockIdx.x * blockDim.x + threadIdx.x; if (i < n) dst[i] = (h16)src[i]; }

// ---------------- head stage 1: TMP[i,o,l] = sum_k h[i,k] Wb[o,k,l] --------
// grid (64 i-tiles, 128 o), 8 waves/block each own 16-wide l tile.
// B fragment from WbT[o][l][k]: contiguous along k.
__global__ __launch_bounds__(256)
void head_tmp_kernel(const h16* __restrict__ hd16, const h16* __restrict__ WbT,
                     h16* __restrict__ TMP)
{
    int wave = threadIdx.x >> 5, lane = threadIdx.x & 31;
    int ml = lane & 15, hf = (lane >= 16);
    int tm = blockIdx.x * 16;     // i tile
    int o  = blockIdx.y;          // 0..127
    int tn = wave * 16;           // l tile
    v8f acc;
    #pragma unroll
    for (int r = 0; r < 8; ++r) acc[r] = 0.0f;
    const h16* Arow = hd16 + (size_t)(tm + ml) * HH + (hf ? 8 : 0);
    const h16* Brow = WbT + ((size_t)o << 14) + (size_t)(tn + ml) * HH + (hf ? 8 : 0);
    #pragma unroll
    for (int kb = 0; kb < 128; kb += 32) {
        v16h af = load_afrag(Arow + kb);
        v16h bf = load_bfrag(Brow + kb);
        acc = __builtin_amdgcn_wmma_f32_16x16x32_f16(false, af, false, bf, (short)0, acc, false, false);
    }
    #pragma unroll
    for (int r = 0; r < 8; ++r) {
        int i = tm + r + (hf ? 8 : 0);
        TMP[((size_t)i * HH + o) * HH + tn + ml] = (h16)acc[r];
    }
}

// ---------------- head stage 2: fused pair + silu-MLP -> logits ------------
// One block per (graph b, node i); 8 waves; each wave owns 2 j-tiles of 16.
// pair tile [16j x 128o] via WMMA, staged in wave-private LDS,
// then m1 = silu(pair@Wm1 + bm1e) via WMMA, then logits = m1@Wm2 + bm2.
__global__ __launch_bounds__(256)
void head_pair_kernel(const h16* __restrict__ hd16, const h16* __restrict__ TMP,
                      const h16* __restrict__ Wm1T, const float* __restrict__ bm1e,
                      const float* __restrict__ Wm2, const float* __restrict__ bm2,
                      float* __restrict__ out)
{
    __shared__ float lds[8][2048];   // 64 KB: per-wave [16 j][128] f32
    int wave = threadIdx.x >> 5, lane = threadIdx.x & 31;
    int ml = lane & 15, hf = (lane >= 16);
    int gi = blockIdx.x;             // 0..1023 = b*256 + i
    int base = (gi >> 8) << 8;       // first node of this graph
    const h16* Tb = TMP + (size_t)gi * HH * HH;
    float* L = lds[wave];

    for (int rep = 0; rep < 2; ++rep) {
        int jt = (wave * 2 + rep) * 16;
        // A fragments: rows of hd (j index), all 4 k-steps (l = 0..127)
        v16h afr[4];
        {
            const h16* Arow = hd16 + (size_t)(base + jt + ml) * HH + (hf ? 8 : 0);
            #pragma unroll
            for (int kb = 0; kb < 4; ++kb) afr[kb] = load_afrag(Arow + kb * 32);
        }
        // pair[j, o] = sum_l hd[j,l] * TMP[i,o,l]
        float pacc[8][8];
        #pragma unroll
        for (int ot = 0; ot < 8; ++ot) {
            v8f acc;
            #pragma unroll
            for (int r = 0; r < 8; ++r) acc[r] = 0.0f;
            const h16* Brow = Tb + (size_t)(ot * 16 + ml) * HH + (hf ? 8 : 0);
            #pragma unroll
            for (int kb = 0; kb < 4; ++kb) {
                v16h bf = load_bfrag(Brow + kb * 32);
                acc = __builtin_amdgcn_wmma_f32_16x16x32_f16(false, afr[kb], false, bf, (short)0, acc, false, false);
            }
            #pragma unroll
            for (int r = 0; r < 8; ++r) pacc[ot][r] = acc[r];
        }
        // stage pair tile to LDS (wave-private, in-order DS)
        #pragma unroll
        for (int ot = 0; ot < 8; ++ot)
            #pragma unroll
            for (int r = 0; r < 8; ++r)
                L[(r + hf * 8) * 128 + ot * 16 + ml] = pacc[ot][r];
        // m1 = pair @ Wm1 + bm1e (WMMA, A from LDS f32->f16, B from Wm1T f16)
        float macc[8][8];
        #pragma unroll
        for (int pt = 0; pt < 8; ++pt) {
            v8f acc;
            float bv = bm1e[pt * 16 + ml];
            #pragma unroll
            for (int r = 0; r < 8; ++r) acc[r] = bv;
            const h16* Brow = Wm1T + (size_t)(pt * 16 + ml) * HH + (hf ? 8 : 0);
            #pragma unroll
            for (int kb = 0; kb < 4; ++kb) {
                v16h af, bf;
                #pragma unroll
                for (int i = 0; i < 16; ++i) {
                    int kk = ((i & 8) ? 16 : 0) + (hf ? 8 : 0) + (i & 7);
                    af[i] = (_Float16)L[ml * 128 + kb * 32 + kk];
                }
                bf = load_bfrag(Brow + kb * 32);
                acc = __builtin_amdgcn_wmma_f32_16x16x32_f16(false, af, false, bf, (short)0, acc, false, false);
            }
            #pragma unroll
            for (int r = 0; r < 8; ++r) macc[pt][r] = acc[r];
        }
        // overwrite LDS with silu(m1)
        #pragma unroll
        for (int pt = 0; pt < 8; ++pt)
            #pragma unroll
            for (int r = 0; r < 8; ++r)
                L[(r + hf * 8) * 128 + pt * 16 + ml] = siluf(macc[pt][r]);
        // logits[j, q] = silu(m1)[j,:] @ Wm2[:,q] + bm2[q]
        #pragma unroll
        for (int oi = lane; oi < 64; oi += 32) {
            int jl = oi >> 2, q = oi & 3;
            float acc = bm2[q];
            for (int p = 0; p < 128; ++p) acc += L[jl * 128 + p] * Wm2[p * 4 + q];
            out[((size_t)gi * NPG + jt + jl) * 4 + q] = acc;
        }
    }
}

// effective bias for m1: bm1 + bb @ Wm1 (reference adds bb to pair; next op
// is linear in pair so it folds exactly).
__global__ void bias_fold_kernel(const float* __restrict__ bb, const float* __restrict__ Wm1,
                                 const float* __restrict__ bm1, float* __restrict__ outb)
{
    int j = threadIdx.x;
    float acc = bm1[j];
    for (int o = 0; o < 128; ++o) acc += bb[o] * Wm1[o * 128 + j];
    outb[j] = acc;
}

// =========================== host launch ===================================
extern "C" void kernel_launch(void* const* d_in, const int* in_sizes, int n_in,
                              void* d_out, int out_size, void* d_ws, size_t ws_size,
                              hipStream_t stream)
{
    const float* x     = (const float*)d_in[0];
    const int*   ei    = (const int*)d_in[1];
    const int*   tptr  = (const int*)d_in[3];
    const float* tmat  = (const float*)d_in[4];
    const float* Wn    = (const float*)d_in[5];
    const float* bn    = (const float*)d_in[6];
    const float* Wee   = (const float*)d_in[7];
    const float* bee   = (const float*)d_in[8];
    const float* Wt1   = (const float*)d_in[9];
    const float* bt1   = (const float*)d_in[10];
    const float* Wt2   = (const float*)d_in[11];
    const float* bt2   = (const float*)d_in[12];
    const float* Wa    = (const float*)d_in[13];
    const float* ba    = (const float*)d_in[14];
    const float* Wq    = (const float*)d_in[15];
    const float* bq    = (const float*)d_in[16];
    const float* Wk    = (const float*)d_in[17];
    const float* bk    = (const float*)d_in[18];
    const float* Wv    = (const float*)d_in[19];
    const float* bv    = (const float*)d_in[20];
    const float* We    = (const float*)d_in[21];
    const float* Wskip = (const float*)d_in[22];
    const float* bskip = (const float*)d_in[23];
    const float* Wf1   = (const float*)d_in[24];
    const float* bf1   = (const float*)d_in[25];
    const float* Wf2   = (const float*)d_in[26];
    const float* bf2   = (const float*)d_in[27];
    const float* Wb    = (const float*)d_in[28];
    const float* bb    = (const float*)d_in[29];
    const float* Wm1   = (const float*)d_in[30];
    const float* bm1   = (const float*)d_in[31];
    const float* Wm2   = (const float*)d_in[32];
    const float* bm2   = (const float*)d_in[33];
    float* out = (float*)d_out;

    float* W = (float*)d_ws;
    size_t off = 0;
    auto alloc = [&](size_t nfloats) { float* p = W + off; off += nfloats; return p; };
    float*    ss     = alloc(NL * 512);
    float*    ea     = alloc(EE);
    float*    h      = alloc(NN * HH);
    float*    qb     = alloc(NN * HH);
    float*    kb     = alloc(NN * HH);
    float*    vb     = alloc(NN * HH);
    float*    skipb  = alloc(NN * HH);
    float*    attnb  = alloc(NN * HH);
    float*    alphab = alloc(EE * 4);
    unsigned* mseg   = (unsigned*)alloc(NN * 4);
    float*    ssum   = alloc(NN * 4);
    float*    f1b    = alloc(NN * 256);
    float*    f2b    = alloc(NN * HH);
    float*    bm1e   = alloc(128);
    h16*      hn16   = (h16*)alloc(NN * HH / 2);
    h16*      f1b16  = (h16*)alloc(NN * 256 / 2);
    h16*      hd16   = (h16*)alloc(NN * HH / 2);
    h16*      wqT    = (h16*)alloc(HH * HH / 2);
    h16*      wkT    = (h16*)alloc(HH * HH / 2);
    h16*      wvT    = (h16*)alloc(HH * HH / 2);
    h16*      wsT    = (h16*)alloc(HH * HH / 2);
    h16*      wf1T   = (h16*)alloc(HH * 256 / 2);
    h16*      wf2T   = (h16*)alloc(256 * HH / 2);
    h16*      Wm1T   = (h16*)alloc(HH * HH / 2);
    h16*      WbT    = (h16*)alloc((size_t)HH * HH * HH / 2);
    h16*      TMP    = (h16*)alloc((size_t)NN * HH * HH / 2);
    (void)ws_size; (void)in_sizes; (void)n_in; (void)out_size;

    setup_kernel<<<1, 512, 0, stream>>>(tptr, Wt1, bt1, Wt2, bt2, Wa, ba, ss);
    input_proj_kernel<<<NN, HH, 0, stream>>>(x, Wn, bn, h);
    edge_kernel<<<EE / 256, 256, 0, stream>>>(ei, tmat, Wee, bee, ea);

    for (int l = 0; l < NL; ++l) {
        const float* shm = ss + l * 512 + 0;    // sh_msa
        const float* scm = ss + l * 512 + 128;  // sc_msa
        const float* shp = ss + l * 512 + 256;  // sh_mlp
        const float* scp = ss + l * 512 + 384;  // sc_mlp
        // transpose+convert this layer's weights to f16 [N,K]
        tconv_kernel<<<(HH * HH) / 256, 256, 0, stream>>>(Wq + l * HH * HH, wqT, HH, HH);
        tconv_kernel<<<(HH * HH) / 256, 256, 0, stream>>>(Wk + l * HH * HH, wkT, HH, HH);
        tconv_kernel<<<(HH * HH) / 256, 256, 0, stream>>>(Wv + l * HH * HH, wvT, HH, HH);
        tconv_kernel<<<(HH * HH) / 256, 256, 0, stream>>>(Wskip + l * HH * HH, wsT, HH, HH);
        tconv_kernel<<<(HH * 256) / 256, 256, 0, stream>>>(Wf1 + l * HH * 256, wf1T, HH, 256);
        tconv_kernel<<<(256 * HH) / 256, 256, 0, stream>>>(Wf2 + l * 256 * HH, wf2T, 256, HH);

        ln_mod_kernel<<<NN, HH, 0, stream>>>(h, shm, scm, hn16);
        dim3 g8(NN / 16, HH / 16);
        gemm_wmma_h16<<<g8, 32, 0, stream>>>(hn16, HH, wqT, bq + l * HH, qb, HH, HH, 0);
        gemm_wmma_h16<<<g8, 32, 0, stream>>>(hn16, HH, wkT, bk + l * HH, kb, HH, HH, 0);
        gemm_wmma_h16<<<g8, 32, 0, stream>>>(hn16, HH, wvT, bv + l * HH, vb, HH, HH, 0);
        gemm_wmma_h16<<<g8, 32, 0, stream>>>(hn16, HH, wsT, bskip + l * HH, skipb, HH, HH, 0);
        layer_init_kernel<<<(NN * HH) / 256, 256, 0, stream>>>(attnb, mseg, ssum);
        const float* WeL = We + l * HH;
        attn_alpha_kernel<<<(EE * 4) / 256, 256, 0, stream>>>(ei, ea, qb, kb, WeL, alphab, mseg);
        attn_exp_kernel<<<(EE * 4) / 256, 256, 0, stream>>>(ei, alphab, mseg, ssum);
        attn_scatter_kernel<<<(EE * 4) / 256, 256, 0, stream>>>(ei, ea, vb, WeL, alphab, ssum, attnb);
        add2_kernel<<<(NN * HH) / 256, 256, 0, stream>>>(h, attnb, skipb);
        ln_mod_kernel<<<NN, HH, 0, stream>>>(h, shp, scp, hn16);
        dim3 g16(NN / 16, 256 / 16);
        gemm_wmma_h16<<<g16, 32, 0, stream>>>(hn16, HH, wf1T, bf1 + l * 256, f1b, 256, HH, 1);
        cvt_kernel<<<(NN * 256) / 256, 256, 0, stream>>>(f1b, f1b16, NN * 256);
        gemm_wmma_h16<<<g8, 32, 0, stream>>>(f1b16, 256, wf2T, bf2 + l * HH, f2b, HH, 256, 0);
        add1_kernel<<<(NN * HH) / 256, 256, 0, stream>>>(h, f2b);
    }

    bias_fold_kernel<<<1, 128, 0, stream>>>(bb, Wm1, bm1, bm1e);
    tconv_kernel<<<(HH * HH) / 256, 256, 0, stream>>>(Wm1, Wm1T, HH, HH);
    tconv_wb_kernel<<<(HH * HH * HH) / 256, 256, 0, stream>>>(Wb, WbT);
    cvt_kernel<<<(NN * HH) / 256, 256, 0, stream>>>(h, hd16, NN * HH);
    head_tmp_kernel<<<dim3(NN / 16, HH), 256, 0, stream>>>(hd16, WbT, TMP);
    head_pair_kernel<<<NN, 256, 0, stream>>>(hd16, TMP, Wm1T, bm1e, Wm2, bm2, out);
}
